// DirectedLowRankEdgeScorer_71055938945737
// MI455X (gfx1250) — matrix operand order repaired and checked by reading, not compiled, not run on hardware
//
#include <hip/hip_runtime.h>

typedef __attribute__((ext_vector_type(16))) _Float16 v16h;
typedef __attribute__((ext_vector_type(8)))  _Float16 v8h;
typedef __attribute__((ext_vector_type(8)))  float    v8f;
typedef __attribute__((ext_vector_type(4)))  float    v4f;
typedef __attribute__((ext_vector_type(2)))  float    v2f;

#define BB   2
#define TT   8
#define NN   50000
#define CC   64
#define HID_ 128
#define RR   16
#define LL   12
#define EE   1600000

#if __has_builtin(__builtin_amdgcn_global_load_async_to_lds_b32) && \
    __has_builtin(__builtin_amdgcn_s_wait_asynccnt)
#define USE_ASYNC_LDS 1
typedef __attribute__((address_space(1))) int gas_int;   // global addrspace
typedef __attribute__((address_space(3))) int las_int;   // LDS addrspace
#else
#define USE_ASYNC_LDS 0
#endif

// ---------------------------------------------------------------------------
// Phase 0: one-shot weight prep. Convert fp32 weights to f16 and transpose to
// column-major so WMMA B-fragments are two contiguous 16B runs per lane:
//   W1T[col][k]  (128 x 64),  W2T[col][k]  (16 x 128)
// ---------------------------------------------------------------------------
__global__ __launch_bounds__(256) void prep_weights_kernel(
    const float* __restrict__ W1s, const float* __restrict__ W1d,
    const float* __restrict__ W2s, const float* __restrict__ W2d,
    _Float16* __restrict__ W1Ts, _Float16* __restrict__ W1Td,
    _Float16* __restrict__ W2Ts, _Float16* __restrict__ W2Td)
{
    const int t = blockIdx.x * 256 + threadIdx.x;
    if (t < CC * HID_) {                       // W1: 64 x 128, row-major
        const int k = t / HID_, c = t % HID_;
        W1Ts[c * CC + k] = (_Float16)W1s[t];
        W1Td[c * CC + k] = (_Float16)W1d[t];
    }
    if (t < HID_ * RR) {                       // W2: 128 x 16, row-major
        const int k = t / RR, c = t % RR;
        W2Ts[c * HID_ + k] = (_Float16)W2s[t];
        W2Td[c * HID_ + k] = (_Float16)W2d[t];
    }
}

// ---------------------------------------------------------------------------
// Phase 1: a = relu(H@W1s+b1s)@W2s+b2s ; b = relu(H@W1d+b1d)@W2d+b2d
// One wave per 16-row tile of H (M = B*N = 100000 rows). f16 WMMA with f32
// accumulation; hidden tile transposed through LDS for the second layer.
// ---------------------------------------------------------------------------
__global__ __launch_bounds__(256) void mlp_fused_kernel(
    const float* __restrict__ X,
    const _Float16* __restrict__ W1Ts, const float* __restrict__ b1s,
    const _Float16* __restrict__ W2Ts, const float* __restrict__ b2s,
    const _Float16* __restrict__ W1Td, const float* __restrict__ b1d,
    const _Float16* __restrict__ W2Td, const float* __restrict__ b2d,
    float* __restrict__ a_tab, float* __restrict__ b_tab)
{
    __shared__ __align__(16) _Float16 lds_h[8][16][HID_];   // 32 KB

    const int tid  = threadIdx.x;
    const int wave = tid >> 5;
    const int lane = tid & 31;
    const int hi16 = lane >> 4;      // 0 for lanes 0-15, 1 for 16-31
    const int lrow = lane & 15;
    const int Mtot = BB * NN;

    const int tile = blockIdx.x * 8 + wave;
    const int row0 = tile * 16;

    // ---- A fragments of H (16x64 fp32 -> two 16x32 f16 fragments) ----
    int growA = row0 + lrow;
    if (growA >= Mtot) growA = Mtot - 1;          // clamp; stores predicated later
    const int rb   = growA / NN;
    const int node = growA - rb * NN;
    const float* Hrow = X + (((size_t)rb * TT + (TT - 1)) * (size_t)NN + (size_t)node) * CC;

    v16h afrag[2];
    #pragma unroll
    for (int ks = 0; ks < 2; ++ks) {
        const int kb = ks * 32 + hi16 * 8;        // ISA 16-bit A layout base
        v4f x0 = *(const v4f*)(Hrow + kb);
        v4f x1 = *(const v4f*)(Hrow + kb + 4);
        v4f x2 = *(const v4f*)(Hrow + kb + 16);
        v4f x3 = *(const v4f*)(Hrow + kb + 20);
        v16h a;
        #pragma unroll
        for (int h = 0; h < 4; ++h) {
            a[h]      = (_Float16)x0[h];
            a[h + 4]  = (_Float16)x1[h];
            a[h + 8]  = (_Float16)x2[h];
            a[h + 12] = (_Float16)x3[h];
        }
        afrag[ks] = a;
    }

    #pragma unroll
    for (int mlp = 0; mlp < 2; ++mlp) {
        const _Float16* W1 = mlp ? W1Td : W1Ts;
        const float*    B1 = mlp ? b1d  : b1s;
        const _Float16* W2 = mlp ? W2Td : W2Ts;
        const float*    B2 = mlp ? b2d  : b2s;
        float*          OT = mlp ? b_tab : a_tab;

        // ---- layer 1: hidden = relu(H @ W1 + b1), 8 column tiles of 16 ----
        #pragma unroll
        for (int nt = 0; nt < 8; ++nt) {
            const int cg = nt * 16 + lrow;        // lane's output column
            v8f acc = {};
            #pragma unroll
            for (int ks = 0; ks < 2; ++ks) {
                // B fragment: two 8-half runs (K = base..+7, base+16..+23) in transposed weights
                const _Float16* wp = W1 + cg * CC + ks * 32 + hi16 * 8;
                v8h w0 = *(const v8h*)wp;
                v8h w1 = *(const v8h*)(wp + 16);
                v16h bf = __builtin_shufflevector(w0, w1,
                              0,1,2,3,4,5,6,7,8,9,10,11,12,13,14,15);
                acc = __builtin_amdgcn_wmma_f32_16x16x32_f16(
                          false, afrag[ks], false, bf, (short)0, acc, false, false);
            }
            const float bias = B1[cg];
            #pragma unroll
            for (int v = 0; v < 8; ++v) {
                float hv = acc[v] + bias;
                hv = hv > 0.f ? hv : 0.f;
                lds_h[wave][v + hi16 * 8][cg] = (_Float16)hv;   // C/D -> row-major
            }
        }
        __syncthreads();

        // ---- layer 2: out = hidden(16x128) @ W2(128x16) + b2 ----
        v8f acc2 = {};
        #pragma unroll
        for (int ks = 0; ks < 4; ++ks) {
            const _Float16* hp = &lds_h[wave][lrow][ks * 32 + hi16 * 8];
            v8h lo = *(const v8h*)hp;              // K = base..base+7
            v8h hh = *(const v8h*)(hp + 16);       // K = base+16..base+23
            v16h a2 = __builtin_shufflevector(lo, hh,
                          0,1,2,3,4,5,6,7,8,9,10,11,12,13,14,15);
            const _Float16* wp2 = W2 + lrow * HID_ + ks * 32 + hi16 * 8;
            v8h u0 = *(const v8h*)wp2;
            v8h u1 = *(const v8h*)(wp2 + 16);
            v16h b2f = __builtin_shufflevector(u0, u1,
                           0,1,2,3,4,5,6,7,8,9,10,11,12,13,14,15);
            acc2 = __builtin_amdgcn_wmma_f32_16x16x32_f16(
                       false, a2, false, b2f, (short)0, acc2, false, false);
        }
        const float bias2 = B2[lrow];
        #pragma unroll
        for (int v = 0; v < 8; ++v) {
            const int grow = row0 + v + hi16 * 8;
            if (grow < Mtot)
                OT[(size_t)grow * RR + lrow] = acc2[v] + bias2;
        }
        __syncthreads();   // lds_h reused by second MLP
    }
}

// ---------------------------------------------------------------------------
// Phase 2: logits[b,l,e] = sum_r a[b,I[e],r] * gamma[l,r] * b[b,J[e],r]
// One wave per two 16-edge tiles. Per tile: P = a_i (.) b_j  (16 edges x 16 r)
// contracted against gamma^T via 4x v_wmma_f32_16x16x4_f32; D tiles staged in
// LDS and written out as coalesced float4 stores (1 KB contiguous per block,l).
// Gamma staged to LDS via the CDNA5 async global->LDS path when available.
// ---------------------------------------------------------------------------
__global__ __launch_bounds__(256) void edge_score_kernel(
    const long long* __restrict__ e64,     // int64 edge indices
    const float* __restrict__ a_tab, const float* __restrict__ b_tab,
    const float* __restrict__ gamma, float* __restrict__ out)
{
    __shared__ float lds_g[256];                        // gamma padded 16x16
    __shared__ __align__(16) float lds_o[8][2][LL][16]; // 12 KB store staging

    const int tid  = threadIdx.x;
    const int wave = tid >> 5;
    const int lane = tid & 31;
    const int hi16 = lane >> 4;
    const int lrow = lane & 15;
    const int bb   = blockIdx.y;

    // stage gamma (row n, col k at lds_g[n*16+k]; rows 12..15 zero-padded)
    if (tid < LL * RR) {
#if USE_ASYNC_LDS
        __builtin_amdgcn_global_load_async_to_lds_b32(
            (gas_int*)(gamma + tid), (las_int*)&lds_g[tid], 0, 0);
#else
        lds_g[tid] = gamma[tid];
#endif
    } else {
        lds_g[tid] = 0.f;
    }
#if USE_ASYNC_LDS
    __builtin_amdgcn_s_wait_asynccnt(0);
#endif
    __syncthreads();

    const size_t be0  = (size_t)blockIdx.x * 256;
    const float* atb  = a_tab + (size_t)bb * NN * RR;
    const float* btb  = b_tab + (size_t)bb * NN * RR;
    const int    off2 = hi16 * 2;          // f32 A/B fragment K offset

    // gamma^T B-fragments depend only on (lane, ks): hoist out of tile loop
    v2f gfr[4];
    #pragma unroll
    for (int ks = 0; ks < 4; ++ks)
        gfr[ks] = *(const v2f*)&lds_g[lrow * 16 + ks * 4 + off2];

    #pragma unroll
    for (int sub = 0; sub < 2; ++sub) {
        const size_t e = be0 + (size_t)wave * 32 + sub * 16 + lrow;
        const int I = (int)e64[e];
        const int J = (int)e64[(size_t)EE + e];
        const float* ar = atb + (size_t)I * RR;
        const float* br = btb + (size_t)J * RR;

        v8f acc = {};
        #pragma unroll
        for (int ks = 0; ks < 4; ++ks) {
            const int rbase = ks * 4 + off2;
            v2f av = *(const v2f*)(ar + rbase);
            v2f bv = *(const v2f*)(br + rbase);
            v2f af; af[0] = av[0] * bv[0]; af[1] = av[1] * bv[1];   // P fragment
            acc = __builtin_amdgcn_wmma_f32_16x16x4_f32(
                      false, af, false, gfr[ks], (short)0, acc, false, false);
        }
        if (lrow < LL) {                   // lane col = l; VGPR v -> edge row
            #pragma unroll
            for (int v = 0; v < 8; ++v)
                lds_o[wave][sub][lrow][v + hi16 * 8] = acc[v];
        }
    }
    __syncthreads();

    // coalesced writeout: 12 layers x 64 float4 chunks = 768 tasks, 3/thread
    #pragma unroll
    for (int it = 0; it < 3; ++it) {
        const int task = it * 256 + tid;
        const int l  = task >> 6;          // 0..11
        const int ei = (task & 63) * 4;    // edge-in-block, 16B aligned
        const int wv = ei >> 5, sb = (ei >> 4) & 1, mm = ei & 15;
        const v4f val = *(const v4f*)&lds_o[wv][sb][l][mm];
        *(v4f*)(out + ((size_t)bb * LL + l) * EE + be0 + ei) = val;
    }
}

extern "C" void kernel_launch(void* const* d_in, const int* in_sizes, int n_in,
                              void* d_out, int out_size, void* d_ws, size_t ws_size,
                              hipStream_t stream)
{
    (void)in_sizes; (void)n_in; (void)out_size; (void)ws_size;
    const float* X     = (const float*)d_in[0];
    const long long* e64 = (const long long*)d_in[1];
    const float* W1s   = (const float*)d_in[2];
    const float* b1s   = (const float*)d_in[3];
    const float* W2s   = (const float*)d_in[4];
    const float* b2s   = (const float*)d_in[5];
    const float* W1d   = (const float*)d_in[6];
    const float* b1d   = (const float*)d_in[7];
    const float* W2d   = (const float*)d_in[8];
    const float* b2d   = (const float*)d_in[9];
    const float* gamma = (const float*)d_in[10];
    float* out = (float*)d_out;

    // workspace layout
    float* a_tab = (float*)d_ws;                              // 6.4 MB
    float* b_tab = a_tab + (size_t)BB * NN * RR;              // 6.4 MB
    _Float16* wbase = (_Float16*)(b_tab + (size_t)BB * NN * RR);
    _Float16* W1Ts = wbase;                                   // 128*64
    _Float16* W1Td = W1Ts + HID_ * CC;
    _Float16* W2Ts = W1Td + HID_ * CC;                        // 16*128
    _Float16* W2Td = W2Ts + RR * HID_;

    prep_weights_kernel<<<(CC * HID_ + 255) / 256, 256, 0, stream>>>(
        W1s, W1d, W2s, W2d, W1Ts, W1Td, W2Ts, W2Td);

    const int ntiles = (BB * NN + 15) / 16;                   // 6250
    dim3 g1((ntiles + 7) / 8);                                // 8 waves/block
    mlp_fused_kernel<<<g1, 256, 0, stream>>>(X, W1Ts, b1s, W2Ts, b2s,
                                             W1Td, b1d, W2Td, b2d, a_tab, b_tab);

    dim3 g2(EE / 256, BB);                                    // 6250 x 2
    edge_score_kernel<<<g2, 256, 0, stream>>>(e64, a_tab, b_tab, gamma, out);
}
